// LSTMCell_9663676416791
// MI455X (gfx1250) — compile-verified, matrix operand order checked by reading, not run on hardware
//
#include <hip/hip_runtime.h>
#include <hip/hip_bf16.h>
#include <cstdint>

// Problem sizes (fixed by the reference)
#define BATCH 16384
#define DH    1024
#define DIN   1024

// Tiling
#define BM 128      // rows per block (8 waves x 16)
#define BN 64       // gate-columns per block (4 WMMA n-tiles)
#define BK 32       // K step (one bf16 WMMA K)
#define LDS_STRIDE 48   // elements per LDS row (96B: 16B-aligned, bank-spread)

typedef __attribute__((ext_vector_type(16))) __bf16 v16bf;
typedef __attribute__((ext_vector_type(8)))  float  v8f;
typedef int  i4v __attribute__((ext_vector_type(4)));

struct Ptrs8 { const float* p[8]; };

// ---------- f32 -> bf16 (round-to-nearest-even), packed pair ----------
__device__ __forceinline__ uint32_t pk_bf16(float a, float b) {
    union { float f; uint32_t u; } ua, ub;
    ua.f = a; ub.f = b;
    uint32_t ra = ua.u + 0x7FFFu + ((ua.u >> 16) & 1u);
    uint32_t rb = ub.u + 0x7FFFu + ((ub.u >> 16) & 1u);
    return (ra >> 16) | (rb & 0xFFFF0000u);
}

// ---------- CDNA5 async global->LDS copy (16B per lane), guarded ----------
#if __has_builtin(__builtin_amdgcn_global_load_async_to_lds_b128)
#define HAVE_ASYNC_LDS 1
#endif

__device__ __forceinline__ void cp16_g2l(void* lds, const void* gmem) {
#ifdef HAVE_ASYNC_LDS
    __builtin_amdgcn_global_load_async_to_lds_b128(
        (__attribute__((address_space(1))) i4v*)(gmem),
        (__attribute__((address_space(3))) i4v*)(lds),
        0, 0);
#else
    *(uint4*)lds = *(const uint4*)gmem;
#endif
}

__device__ __forceinline__ void async_wait_all() {
#ifdef HAVE_ASYNC_LDS
#if __has_builtin(__builtin_amdgcn_s_wait_asynccnt)
    __builtin_amdgcn_s_wait_asynccnt(0);
#else
    asm volatile("s_wait_asynccnt 0" ::: "memory");
#endif
#endif
}

// ---------- load one 16x32 bf16 fragment from an LDS tile ----------
// CDNA5 16-bit A/B layout: lanes 0-15 hold rows 0-15 with K=0..7,16..23;
// lanes 16-31 hold rows 0-15 with K=8..15,24..31.
__device__ __forceinline__ v16bf load_frag16x32(const __bf16* tile, int lane) {
    const int r  = lane & 15;
    const int kh = (lane >> 4) << 3;           // 0 or 8
    const uint32_t* p0 = (const uint32_t*)(tile + r * LDS_STRIDE + kh);
    const uint32_t* p1 = (const uint32_t*)(tile + r * LDS_STRIDE + kh + 16);
    union { v16bf v; uint32_t d[8]; } u;
    u.d[0] = p0[0]; u.d[1] = p0[1]; u.d[2] = p0[2]; u.d[3] = p0[3];
    u.d[4] = p1[0]; u.d[5] = p1[1]; u.d[6] = p1[2]; u.d[7] = p1[3];
    return u.v;
}

__device__ __forceinline__ float fast_sigmoid(float x) {
    return 1.0f / (1.0f + __expf(-x));
}

// =====================================================================
// Kernel 1: convert the 8 f32 weight matrices to bf16 in workspace.
// Layout: ws[(g*DH + row)*DIN + k], g = {W_i,W_o,W_f,W_c,U_i,U_o,U_f,U_c}
// =====================================================================
__global__ __launch_bounds__(256) void convert_weights_kernel(Ptrs8 w, uint16_t* __restrict__ out) {
    const int g = blockIdx.y;
    const float* __restrict__ src = w.p[g];
    const size_t off = (size_t)blockIdx.x * (256 * 8) + (size_t)threadIdx.x * 8;
    float4 a = *(const float4*)(src + off);
    float4 b = *(const float4*)(src + off + 4);
    uint4 o;
    o.x = pk_bf16(a.x, a.y); o.y = pk_bf16(a.z, a.w);
    o.z = pk_bf16(b.x, b.y); o.w = pk_bf16(b.z, b.w);
    *(uint4*)(out + (size_t)g * DH * DIN + off) = o;
}

// =====================================================================
// Kernel 2: fused LSTM cell.
// Each block: 128 rows x 64 gate-cols, all 4 gates, full K reduction
// over X@W^T and h@U^T, then the elementwise cell update + stores.
// =====================================================================
__global__ __launch_bounds__(256) void lstm_fused_kernel(
    const float* __restrict__ X, const float* __restrict__ Hin, const float* __restrict__ Cin,
    const uint16_t* __restrict__ Wbf,    // [8][DH][DIN] bf16 bits
    Ptrs8 bias,                          // bWi,bWo,bWf,bWc,bUi,bUo,bUf,bUc
    float* __restrict__ out)
{
    __shared__ __align__(16) __bf16 ldsX[BM * LDS_STRIDE];
    __shared__ __align__(16) __bf16 ldsH[BM * LDS_STRIDE];
    __shared__ __align__(16) __bf16 ldsW[4][BN * LDS_STRIDE];
    __shared__ __align__(16) __bf16 ldsU[4][BN * LDS_STRIDE];

    const int t    = threadIdx.x;
    const int lane = t & 31;
    const int wave = t >> 5;                 // 0..7, each owns 16 rows
    const int mBase = blockIdx.y * BM;
    const int nBase = blockIdx.x * BN;

    v8f acc[4][4];                            // [gate][n-tile]
#pragma unroll
    for (int g = 0; g < 4; ++g)
#pragma unroll
        for (int nt = 0; nt < 4; ++nt)
#pragma unroll
            for (int r = 0; r < 8; ++r) acc[g][nt][r] = 0.0f;

    // staging maps (256 threads)
    const int rowX = t >> 1;                  // 0..127
    const int colX = (t & 1) * 16;            // 0 or 16 (8 floats each)
    const int rowW = t >> 2;                  // 0..63
    const int chkW = (t & 3) * 8;             // 0,8,16,24 elements (16B)

    for (int k0 = 0; k0 < DIN; k0 += BK) {
        // ---- stage X and h tiles: f32 load, cvt to bf16, LDS store ----
        {
            const float4* px = (const float4*)(X   + (size_t)(mBase + rowX) * DIN + k0 + colX);
            const float4* ph = (const float4*)(Hin + (size_t)(mBase + rowX) * DH  + k0 + colX);
            float4 x0 = px[0], x1 = px[1];
            float4 h0 = ph[0], h1 = ph[1];
            uint32_t* dx = (uint32_t*)(ldsX + rowX * LDS_STRIDE + colX);
            uint32_t* dh = (uint32_t*)(ldsH + rowX * LDS_STRIDE + colX);
            dx[0] = pk_bf16(x0.x, x0.y); dx[1] = pk_bf16(x0.z, x0.w);
            dx[2] = pk_bf16(x1.x, x1.y); dx[3] = pk_bf16(x1.z, x1.w);
            dh[0] = pk_bf16(h0.x, h0.y); dh[1] = pk_bf16(h0.z, h0.w);
            dh[2] = pk_bf16(h1.x, h1.y); dh[3] = pk_bf16(h1.z, h1.w);
        }
        // ---- stage bf16 weight tiles via async global->LDS (16B/lane) ----
#pragma unroll
        for (int g = 0; g < 4; ++g) {
            const uint16_t* gw = Wbf + ((size_t)g       * DH + nBase + rowW) * DIN + k0 + chkW;
            const uint16_t* gu = Wbf + ((size_t)(g + 4) * DH + nBase + rowW) * DIN + k0 + chkW;
            cp16_g2l(ldsW[g] + rowW * LDS_STRIDE + chkW, gw);
            cp16_g2l(ldsU[g] + rowW * LDS_STRIDE + chkW, gu);
        }
        async_wait_all();
        __syncthreads();

        // ---- WMMA: 32 v_wmma_f32_16x16x32_bf16 per K-step per wave ----
        const v16bf aX = load_frag16x32(ldsX + wave * 16 * LDS_STRIDE, lane);
        const v16bf aH = load_frag16x32(ldsH + wave * 16 * LDS_STRIDE, lane);
#pragma unroll
        for (int g = 0; g < 4; ++g) {
#pragma unroll
            for (int nt = 0; nt < 4; ++nt) {
                v16bf bW = load_frag16x32(ldsW[g] + nt * 16 * LDS_STRIDE, lane);
                acc[g][nt] = __builtin_amdgcn_wmma_f32_16x16x32_bf16(
                    false, aX, false, bW, (short)0, acc[g][nt], false, false);
                v16bf bU = load_frag16x32(ldsU[g] + nt * 16 * LDS_STRIDE, lane);
                acc[g][nt] = __builtin_amdgcn_wmma_f32_16x16x32_bf16(
                    false, aH, false, bU, (short)0, acc[g][nt], false, false);
            }
        }
        __syncthreads();
    }

    // ---- epilogue: bias + sigmoid/tanh + cell update + stores ----
    // C/D layout: VGPR r, lanes 0-15 -> (M=r, N=lane); lanes 16-31 -> (M=r+8, N=lane-16)
    const int   m0   = mBase + wave * 16 + ((lane >> 4) << 3);
    const int   nIdx = lane & 15;
    const size_t BD  = (size_t)BATCH * DH;

#pragma unroll
    for (int nt = 0; nt < 4; ++nt) {
        const int col = nBase + nt * 16 + nIdx;
        const float bI = bias.p[0][col] + bias.p[4][col];
        const float bO = bias.p[1][col] + bias.p[5][col];
        const float bF = bias.p[2][col] + bias.p[6][col];
        const float bC = bias.p[3][col] + bias.p[7][col];
#pragma unroll
        for (int r = 0; r < 8; ++r) {
            const size_t idx = (size_t)(m0 + r) * DH + col;
            const float iv = fast_sigmoid(acc[0][nt][r] + bI);
            const float ov = fast_sigmoid(acc[1][nt][r] + bO);
            const float fv = fast_sigmoid(acc[2][nt][r] + bF);
            const float ch = acc[3][nt][r] + bC;
            const float cn = fv * Cin[idx] + iv * ch;
            const float hn = ov * tanhf(cn);
            out[idx]          = hn;   // H (first tuple element)
            out[BD + idx]     = hn;   // H (inside nested tuple)
            out[2 * BD + idx] = cn;   // C
        }
    }
}

// =====================================================================
// Host-side launcher
// =====================================================================
extern "C" void kernel_launch(void* const* d_in, const int* in_sizes, int n_in,
                              void* d_out, int out_size, void* d_ws, size_t ws_size,
                              hipStream_t stream) {
    (void)in_sizes; (void)n_in; (void)out_size; (void)ws_size;

    const float* X   = (const float*)d_in[0];
    const float* Hin = (const float*)d_in[1];
    const float* Cin = (const float*)d_in[2];

    // dict order: W_i,b_W_i, W_o,b_W_o, W_f,b_W_f, W_c,b_W_c, U_i,b_U_i, ...
    Ptrs8 weights = {{ (const float*)d_in[3],  (const float*)d_in[5],
                       (const float*)d_in[7],  (const float*)d_in[9],
                       (const float*)d_in[11], (const float*)d_in[13],
                       (const float*)d_in[15], (const float*)d_in[17] }};
    Ptrs8 biases  = {{ (const float*)d_in[4],  (const float*)d_in[6],
                       (const float*)d_in[8],  (const float*)d_in[10],
                       (const float*)d_in[12], (const float*)d_in[14],
                       (const float*)d_in[16], (const float*)d_in[18] }};

    uint16_t* wbf = (uint16_t*)d_ws;   // 8 * 1024 * 1024 * 2B = 16 MB

    // Pass 1: f32 -> bf16 weights
    convert_weights_kernel<<<dim3((DH * DIN) / (256 * 8), 8), 256, 0, stream>>>(weights, wbf);

    // Pass 2: fused LSTM cell
    lstm_fused_kernel<<<dim3(DH / BN, BATCH / BM), 256, 0, stream>>>(
        X, Hin, Cin, wbf, biases, (float*)d_out);
}
